// Prompt_CrossAttention_57698590654517
// MI455X (gfx1250) — matrix operand order repaired and checked by reading, not compiled, not run on hardware
//
#include <hip/hip_runtime.h>
#include <hip/hip_bf16.h>

// ---------------- problem constants ----------------
#define NB   8          // batch
#define LP   64
#define LS   512
#define LL   576        // LP+LS
#define DD   768
#define VV   50257
#define NH   12
#define HD   64         // head dim
#define M1   (NB*LL)    // 4608 rows of x
#define N1   (3*DD)     // 2304
#define TOPK 10

typedef __attribute__((ext_vector_type(16))) __bf16 v16bf;
typedef __attribute__((ext_vector_type(8)))  float  v8f;

union FragBF { v16bf v; float4 f4[2]; };

__device__ __forceinline__ unsigned short f2bf(float f) {
    unsigned int u = __builtin_bit_cast(unsigned int, f);
    unsigned int r = u + 0x7FFFu + ((u >> 16) & 1u);   // RNE
    return (unsigned short)(r >> 16);
}

// A-matrix fragment (16x32 bf16, ISA 7.12.2): lane row = lane&15,
// elems 0..7 -> K = kk + 8*(lane>>4) + e ; elems 8..15 -> +16.
__device__ __forceinline__ FragBF load_fragA(const unsigned short* base, size_t lda,
                                             int m0, int kk, int lane) {
    int half = lane >> 4, lr = lane & 15;
    const unsigned short* p = base + (size_t)(m0 + lr) * lda + kk + 8 * half;
    FragBF a;
    a.f4[0] = *(const float4*)(p);
    a.f4[1] = *(const float4*)(p + 16);
    return a;
}
// B-matrix fragment (32x16 bf16) from row-major NxK matrix:
// lane col = lane&15, elems 0..7 -> K = kk + 8*(lane>>4) + e ; elems 8..15 -> +16.
__device__ __forceinline__ FragBF load_fragB(const unsigned short* base, size_t ldb,
                                             int n0, int kk, int lane) {
    int half = lane >> 4, lr = lane & 15;
    const unsigned short* p = base + (size_t)(n0 + lr) * ldb + kk + 8 * half;
    FragBF b;
    b.f4[0] = *(const float4*)(p);
    b.f4[1] = *(const float4*)(p + 16);
    return b;
}

#define WMMA_BF16(a, b, c) \
    __builtin_amdgcn_wmma_f32_16x16x32_bf16(false, (a).v, false, (b).v, (short)0, (c), false, false)

// ---------------- cast kernels ----------------
__global__ __launch_bounds__(256) void k_cast_x(const float* __restrict__ prompt,
                                                const float* __restrict__ enc,
                                                unsigned short* __restrict__ Xb) {
    int t = blockIdx.x * 256 + threadIdx.x;
    if (t >= M1 * DD) return;
    int r = t / DD, d = t % DD;
    int b = r / LL, l = r % LL;
    float v = (l < LP) ? prompt[((size_t)b * LP + l) * DD + d]
                       : enc[((size_t)b * LS + (l - LP)) * DD + d];
    Xb[t] = f2bf(v);
}

__global__ __launch_bounds__(256) void k_cast_w(const float* __restrict__ src,
                                                unsigned short* __restrict__ dst, int n) {
    int t = blockIdx.x * 256 + threadIdx.x;
    if (t < n) dst[t] = f2bf(src[t]);
}

// scatter-store one 16x16 fp32 accumulator into Q/K/Vt bf16 buffers
__device__ __forceinline__ void store_qkv_tile(const v8f& acc, int m0, int n0,
                                               int half, int lr,
                                               unsigned short* __restrict__ Qb,
                                               unsigned short* __restrict__ Kb,
                                               unsigned short* __restrict__ Vtb) {
    int part = n0 / DD;
    int h    = (n0 % DD) / HD;
    int d0   = n0 % HD;
#pragma unroll
    for (int r = 0; r < 8; ++r) {
        int m = m0 + r + 8 * half;
        int bb = m / LL, l = m % LL;
        int d = d0 + lr;
        unsigned short val = f2bf(acc[r]);
        size_t bh = (size_t)bb * NH + h;
        if (part == 0)      Qb [(bh * LL + l) * HD + d] = val;
        else if (part == 1) Kb [(bh * LL + l) * HD + d] = val;
        else                Vtb[(bh * HD + d) * LL + l] = val;   // transposed V
    }
}

// ---------------- GEMM1: qkv = X @ Wqkv^T, 32x32 per wave, -> Q/K/Vt bf16 ----------------
__global__ __launch_bounds__(256) void k_gemm_qkv(const unsigned short* __restrict__ Xb,
                                                  const unsigned short* __restrict__ Wb,
                                                  unsigned short* __restrict__ Qb,
                                                  unsigned short* __restrict__ Kb,
                                                  unsigned short* __restrict__ Vtb) {
    const int NT = N1 / 32;                        // 72
    int wave = threadIdx.x >> 5, lane = threadIdx.x & 31;
    int tile = blockIdx.x * 8 + wave;
    if (tile >= (M1 / 32) * NT) return;
    int mt = tile / NT, nt = tile % NT;
    int m0 = mt * 32, n0 = nt * 32;
    int half = lane >> 4, lr = lane & 15;

    v8f c00 = {0.f,0.f,0.f,0.f,0.f,0.f,0.f,0.f};
    v8f c01 = c00, c10 = c00, c11 = c00;
    for (int kk = 0; kk < DD; kk += 32) {
        FragBF a0 = load_fragA(Xb, DD, m0,      kk, lane);
        FragBF a1 = load_fragA(Xb, DD, m0 + 16, kk, lane);
        FragBF b0 = load_fragB(Wb, DD, n0,      kk, lane);
        FragBF b1 = load_fragB(Wb, DD, n0 + 16, kk, lane);
        c00 = WMMA_BF16(a0, b0, c00);
        c01 = WMMA_BF16(a0, b1, c01);
        c10 = WMMA_BF16(a1, b0, c10);
        c11 = WMMA_BF16(a1, b1, c11);
    }
    store_qkv_tile(c00, m0,      n0,      half, lr, Qb, Kb, Vtb);
    store_qkv_tile(c01, m0,      n0 + 16, half, lr, Qb, Kb, Vtb);
    store_qkv_tile(c10, m0 + 16, n0,      half, lr, Qb, Kb, Vtb);
    store_qkv_tile(c11, m0 + 16, n0 + 16, half, lr, Qb, Kb, Vtb);
}

// ---------------- fused attention per (b,h, 16-row tile) ----------------
__global__ __launch_bounds__(256) void k_attn(const unsigned short* __restrict__ Qb,
                                              const unsigned short* __restrict__ Kb,
                                              const unsigned short* __restrict__ Vtb,
                                              unsigned short* __restrict__ AOb) {
    __shared__ __align__(16) float          S[16][580];
    __shared__ __align__(16) unsigned short P[16][584];
    __shared__ float red[16][16];
    __shared__ float rowstat[16];
    __shared__ float Oacc[2][16][HD];

    int bh = blockIdx.x / (LL / 16);
    int mt = blockIdx.x % (LL / 16);
    int bb = bh / NH, h = bh % NH;
    const unsigned short* Qh = Qb  + (size_t)bh * LL * HD;
    const unsigned short* Kh = Kb  + (size_t)bh * LL * HD;
    const unsigned short* Vh = Vtb + (size_t)bh * HD * LL;

    int wave = threadIdx.x >> 5, lane = threadIdx.x & 31;
    int half = lane >> 4, lr = lane & 15;
    int m0 = mt * 16;

    // Phase 1: S[16][576] = (Q K^T) * scale — Q fragments hoisted, tiles over 8 waves
    {
        FragBF qa0 = load_fragA(Qh, HD, m0, 0,  lane);
        FragBF qa1 = load_fragA(Qh, HD, m0, 32, lane);
        for (int jt = wave; jt < LL / 16; jt += 8) {
            v8f acc = {0.f,0.f,0.f,0.f,0.f,0.f,0.f,0.f};
            FragBF b0 = load_fragB(Kh, HD, jt * 16, 0,  lane);
            FragBF b1 = load_fragB(Kh, HD, jt * 16, 32, lane);
            acc = WMMA_BF16(qa0, b0, acc);
            acc = WMMA_BF16(qa1, b1, acc);
#pragma unroll
            for (int r = 0; r < 8; ++r)
                S[r + 8 * half][jt * 16 + lr] = acc[r] * 0.125f;   // 1/sqrt(64)
        }
    }
    __syncthreads();

    // Phase 2: softmax rows (16 threads per row), P = bf16(softmax(S))
    int srow = threadIdx.x >> 4, sl = threadIdx.x & 15;
    float mx = -1e30f;
    for (int c = sl; c < LL; c += 16) mx = fmaxf(mx, S[srow][c]);
    red[srow][sl] = mx;
    __syncthreads();
    if (sl == 0) {
        float m = -1e30f;
#pragma unroll
        for (int i = 0; i < 16; ++i) m = fmaxf(m, red[srow][i]);
        rowstat[srow] = m;
    }
    __syncthreads();
    float rm = rowstat[srow];
    float sum = 0.f;
    for (int c = sl; c < LL; c += 16) {
        float e = __expf(S[srow][c] - rm);
        S[srow][c] = e;
        sum += e;
    }
    red[srow][sl] = sum;
    __syncthreads();
    if (sl == 0) {
        float s = 0.f;
#pragma unroll
        for (int i = 0; i < 16; ++i) s += red[srow][i];
        rowstat[srow] = s;
    }
    __syncthreads();
    float inv = 1.0f / rowstat[srow];
    for (int c = sl; c < LL; c += 16) P[srow][c] = f2bf(S[srow][c] * inv);
    __syncthreads();

    // Phase 3: O = P @ V — all 8 waves: 4 n-tiles x 2 K-halves (288 each), partials in LDS
    {
        int nt    = wave & 3;
        int khalf = wave >> 2;
        int n0    = nt * 16;
        int kbeg  = khalf * (LL / 2);
        v8f acc = {0.f,0.f,0.f,0.f,0.f,0.f,0.f,0.f};
        for (int kk = kbeg; kk < kbeg + LL / 2; kk += 32) {
            FragBF a;
            const unsigned short* ap = &P[lr][kk + 8 * half];
            a.f4[0] = *(const float4*)(ap);
            a.f4[1] = *(const float4*)(ap + 16);
            FragBF b = load_fragB(Vh, LL, n0, kk, lane);
            acc = WMMA_BF16(a, b, acc);
        }
#pragma unroll
        for (int r = 0; r < 8; ++r)
            Oacc[khalf][r + 8 * half][n0 + lr] = acc[r];
    }
    __syncthreads();
    // combine halves and store bf16
    for (int idx = threadIdx.x; idx < 16 * HD; idx += 256) {
        int m = idx / HD, d = idx % HD;
        float v = Oacc[0][m][d] + Oacc[1][m][d];
        AOb[((size_t)bb * LL + m0 + m) * DD + h * HD + d] = f2bf(v);
    }
}

// ---------------- GEMM2: POut = AO @ Wproj^T + bproj, 32x32 per wave (fp32 out) ----------------
__global__ __launch_bounds__(256) void k_gemm_proj(const unsigned short* __restrict__ AOb,
                                                   const unsigned short* __restrict__ Wb,
                                                   const float* __restrict__ bias,
                                                   float* __restrict__ POut) {
    const int NT = DD / 32;                        // 24
    int wave = threadIdx.x >> 5, lane = threadIdx.x & 31;
    int tile = blockIdx.x * 8 + wave;
    if (tile >= (M1 / 32) * NT) return;
    int mt = tile / NT, nt = tile % NT;
    int m0 = mt * 32, n0 = nt * 32;
    int half = lane >> 4, lr = lane & 15;

    v8f c00 = {0.f,0.f,0.f,0.f,0.f,0.f,0.f,0.f};
    v8f c01 = c00, c10 = c00, c11 = c00;
    for (int kk = 0; kk < DD; kk += 32) {
        FragBF a0 = load_fragA(AOb, DD, m0,      kk, lane);
        FragBF a1 = load_fragA(AOb, DD, m0 + 16, kk, lane);
        FragBF b0 = load_fragB(Wb,  DD, n0,      kk, lane);
        FragBF b1 = load_fragB(Wb,  DD, n0 + 16, kk, lane);
        c00 = WMMA_BF16(a0, b0, c00);
        c01 = WMMA_BF16(a0, b1, c01);
        c10 = WMMA_BF16(a1, b0, c10);
        c11 = WMMA_BF16(a1, b1, c11);
    }
    float bv0 = bias[n0 + lr];
    float bv1 = bias[n0 + 16 + lr];
#pragma unroll
    for (int r = 0; r < 8; ++r) {
        int mA = m0 + r + 8 * half;
        int mB = m0 + 16 + r + 8 * half;
        POut[(size_t)mA * DD + n0 + lr]      = c00[r] + bv0;
        POut[(size_t)mA * DD + n0 + 16 + lr] = c01[r] + bv1;
        POut[(size_t)mB * DD + n0 + lr]      = c10[r] + bv0;
        POut[(size_t)mB * DD + n0 + 16 + lr] = c11[r] + bv1;
    }
}

// ---------------- column sum over tokens: s[b][d] = sum_l POut[b,l,d] (float4) ----------------
__global__ __launch_bounds__(256) void k_colsum(const float* __restrict__ POut,
                                                float* __restrict__ s) {
    int t = blockIdx.x * 256 + threadIdx.x;
    if (t >= NB * DD / 4) return;
    int b = t / (DD / 4), d4 = t % (DD / 4);
    const float4* p = (const float4*)(POut + (size_t)b * LL * DD) + d4;
    float4 acc = {0.f, 0.f, 0.f, 0.f};
    for (int l = 0; l < LL; ++l) {
        float4 v = p[(size_t)l * (DD / 4)];
        acc.x += v.x; acc.y += v.y; acc.z += v.z; acc.w += v.w;
    }
    ((float4*)s)[t] = acc;
}

// ---------------- sim[b][v] = dot(s[b], word_emd[v]) — wave per vocab row, b128 loads --------
__global__ __launch_bounds__(256) void k_sim(const float* __restrict__ word_emd,
                                             const float* __restrict__ s,
                                             float* __restrict__ sim) {
    __shared__ __align__(16) float sl[NB * DD];
    for (int i = threadIdx.x; i < NB * DD; i += 256) sl[i] = s[i];
    __syncthreads();

    int gw   = (blockIdx.x * 256 + threadIdx.x) >> 5;
    int lane = threadIdx.x & 31;
    int nw   = (gridDim.x * 256) >> 5;
    for (int v = gw; v < VV; v += nw) {
        const float4* wrow = (const float4*)(word_emd + (size_t)v * DD);
        if (v + nw < VV)
            __builtin_prefetch(word_emd + (size_t)(v + nw) * DD + lane * 4, 0, 1);
        float a[NB] = {0.f,0.f,0.f,0.f,0.f,0.f,0.f,0.f};
#pragma unroll
        for (int i = lane; i < DD / 4; i += 32) {        // 6 x b128 per lane
            float4 w = wrow[i];
#pragma unroll
            for (int b = 0; b < NB; ++b) {
                float4 sv = *(const float4*)&sl[b * DD + i * 4];
                a[b] += w.x * sv.x + w.y * sv.y + w.z * sv.z + w.w * sv.w;
            }
        }
#pragma unroll
        for (int b = 0; b < NB; ++b) {
            a[b] += __shfl_xor(a[b], 16, 32);
            a[b] += __shfl_xor(a[b],  8, 32);
            a[b] += __shfl_xor(a[b],  4, 32);
            a[b] += __shfl_xor(a[b],  2, 32);
            a[b] += __shfl_xor(a[b],  1, 32);
        }
        if (lane == 0) {
#pragma unroll
            for (int b = 0; b < NB; ++b) sim[(size_t)b * VV + v] = a[b];
        }
    }
}

// ---------------- per-batch top-10 ----------------
__global__ __launch_bounds__(256) void k_topk(const float* __restrict__ sim,
                                              int* __restrict__ topk) {
    __shared__ float cv[256 * TOPK];
    __shared__ int   ci[256 * TOPK];
    int b = blockIdx.x, t = threadIdx.x;
    float bv[TOPK];
    int   bi[TOPK];
#pragma unroll
    for (int k = 0; k < TOPK; ++k) { bv[k] = -1e30f; bi[k] = 0; }
    const float* row = sim + (size_t)b * VV;
    for (int v = t; v < VV; v += 256) {
        float x = row[v];
        if (x > bv[TOPK - 1]) {
            int j = TOPK - 1;
            while (j > 0 && bv[j - 1] < x) { bv[j] = bv[j - 1]; bi[j] = bi[j - 1]; --j; }
            bv[j] = x; bi[j] = v;
        }
    }
#pragma unroll
    for (int k = 0; k < TOPK; ++k) { cv[t * TOPK + k] = bv[k]; ci[t * TOPK + k] = bi[k]; }
    __syncthreads();
    if (t == 0) {
        for (int k = 0; k < TOPK; ++k) {
            float best = -1e30f; int arg = 0;
            for (int i = 0; i < 256 * TOPK; ++i)
                if (cv[i] > best) { best = cv[i]; arg = i; }
            topk[b * TOPK + k] = ci[arg];
            cv[arg] = -1e30f;
        }
    }
}

// ---------------- gather rows of word_emd ----------------
__global__ __launch_bounds__(256) void k_gather(const float* __restrict__ word_emd,
                                                const int* __restrict__ topk,
                                                float* __restrict__ out) {
    int t = blockIdx.x * 256 + threadIdx.x;
    if (t >= NB * TOPK * DD) return;
    int d  = t % DD;
    int bk = t / DD;                 // b*TOPK + k
    out[t] = word_emd[(size_t)topk[bk] * DD + d];
}

// ---------------- host launcher ----------------
static inline size_t alignup(size_t x) { return (x + 255) & ~(size_t)255; }

extern "C" void kernel_launch(void* const* d_in, const int* in_sizes, int n_in,
                              void* d_out, int out_size, void* d_ws, size_t ws_size,
                              hipStream_t stream) {
    (void)in_sizes; (void)n_in; (void)out_size; (void)ws_size;
    const float* prompt   = (const float*)d_in[0];
    const float* enc_emd  = (const float*)d_in[1];
    const float* word_emd = (const float*)d_in[2];
    const float* Wqkv     = (const float*)d_in[3];
    const float* Wproj    = (const float*)d_in[4];
    const float* bproj    = (const float*)d_in[5];
    float* out = (float*)d_out;

    char* ws = (char*)d_ws;
    size_t off = 0;
    auto take = [&](size_t bytes) { char* p = ws + off; off += alignup(bytes); return p; };

    unsigned short* Xb     = (unsigned short*)take((size_t)M1 * DD * 2);
    unsigned short* Wqkvb  = (unsigned short*)take((size_t)N1 * DD * 2);
    unsigned short* Wprojb = (unsigned short*)take((size_t)DD * DD * 2);
    unsigned short* Qb     = (unsigned short*)take((size_t)NB * NH * LL * HD * 2);
    unsigned short* Kb     = (unsigned short*)take((size_t)NB * NH * LL * HD * 2);
    unsigned short* Vtb    = (unsigned short*)take((size_t)NB * NH * HD * LL * 2);
    unsigned short* AOb    = (unsigned short*)take((size_t)M1 * DD * 2);
    float*          POut   = (float*)take((size_t)M1 * DD * 4);
    float*          sbuf   = (float*)take((size_t)NB * DD * 4);
    float*          sim    = (float*)take((size_t)NB * VV * 4);
    int*            topk   = (int*)take((size_t)NB * TOPK * 4);

    // 1) casts
    k_cast_x<<<(M1 * DD + 255) / 256, 256, 0, stream>>>(prompt, enc_emd, Xb);
    k_cast_w<<<(N1 * DD + 255) / 256, 256, 0, stream>>>(Wqkv, Wqkvb, N1 * DD);
    k_cast_w<<<(DD * DD + 255) / 256, 256, 0, stream>>>(Wproj, Wprojb, DD * DD);

    // 2) qkv GEMM (WMMA bf16, 32x32/wave) -> Q/K/Vt
    k_gemm_qkv<<<(M1 / 32) * (N1 / 32) / 8, 256, 0, stream>>>(Xb, Wqkvb, Qb, Kb, Vtb);

    // 3) fused attention
    k_attn<<<NB * NH * (LL / 16), 256, 0, stream>>>(Qb, Kb, Vtb, AOb);

    // 4) projection GEMM (WMMA bf16, 32x32/wave, fp32 out + bias)
    k_gemm_proj<<<(M1 / 32) * (DD / 32) / 8, 256, 0, stream>>>(AOb, Wprojb, bproj, POut);

    // 5) token-sum, 6) vocab similarity, 7) top-k, 8) gather
    k_colsum<<<(NB * DD / 4 + 255) / 256, 256, 0, stream>>>(POut, sbuf);
    k_sim<<<512, 256, 0, stream>>>(word_emd, sbuf, sim);
    k_topk<<<NB, 256, 0, stream>>>(sim, topk);
    k_gather<<<(NB * TOPK * DD + 255) / 256, 256, 0, stream>>>(word_emd, topk, out);
}
